// A2HNet_GAT_1013612282043
// MI455X (gfx1250) — matrix operand order repaired and checked by reading, not compiled, not run on hardware
//
#include <hip/hip_runtime.h>
#include <hip/hip_bf16.h>

typedef __attribute__((ext_vector_type(16))) _Float16 v16h;
typedef __attribute__((ext_vector_type(8)))  _Float16 v8h;
typedef __attribute__((ext_vector_type(4)))  _Float16 v4h;
typedef __attribute__((ext_vector_type(8)))  float    v8f;

// ---------------- monotonic float<->uint mapping for atomic float max ----------------
__device__ __forceinline__ unsigned ord_map(float f) {
  unsigned u = __float_as_uint(f);
  return (u & 0x80000000u) ? ~u : (u | 0x80000000u);
}
__device__ __forceinline__ float ord_unmap(unsigned u) {
  return __uint_as_float((u & 0x80000000u) ? (u & 0x7fffffffu) : ~u);
}

// ---------------- generic WMMA GEMM (f16 inputs, f32 accumulate) ----------------
// C[M,N] (ldC row-major) = act( beta*C + A*B + bias )
// A row r lives at A + rowoff(r), rowoff = (r/groupQ)*groupS + (r%groupQ)*ldA
// (groupQ<=0 -> plain r*ldA). B element [k,n] at B[k*sBk + n*sBn].
// act: 0=none 1=relu 2=elu ; epilogue (bias/act) applied only when epi!=0.
// Block tile: 64(M) x 32(N) x 32(K); 4 waves; wave w owns rows w*16..w*16+15,
// computing two 16x16 WMMA tiles (N halves) reusing one A fragment.
#define BM 64
#define BN 32
#define BK 32
__global__ __launch_bounds__(128)
void gemm_wmma(const float* __restrict__ A, int ldA, int groupQ, long long groupS,
               const float* __restrict__ B, long long sBk, long long sBn,
               float* __restrict__ C, int ldC,
               int M, int N, int K,
               const float* __restrict__ bias, int act, int beta, int epi)
{
  __shared__ _Float16 As[BM][BK + 8];     // [row][k]
  __shared__ _Float16 Bs[BN][BK + 8];     // transposed: [n][k]  (fragment-native)
  __shared__ long long rowOff[BM];

  const int tid  = threadIdx.x;
  const int lane = tid & 31;
  const int wave = tid >> 5;
  const int rowBase = blockIdx.y * BM;
  const int colBase = blockIdx.x * BN;

  // hoist grouped-row offsets: computed once per block (kills per-element division)
  for (int r = tid; r < BM; r += 128) {
    int gr = rowBase + r;
    long long off = -1;
    if (gr < M) {
      off = (groupQ > 0)
        ? (long long)(gr / groupQ) * groupS + (long long)(gr % groupQ) * (long long)ldA
        : (long long)gr * (long long)ldA;
    }
    rowOff[r] = off;
  }
  __syncthreads();

  v8f acc0 = {}, acc1 = {};
  const int mlane = lane & 15;
  const int hi    = lane >> 4;

  for (int k0 = 0; k0 < K; k0 += BK) {
    // ---- stage A tile: 64x32, one float4 (b128) per thread-iter, b64 LDS store
    for (int idx = tid; idx < BM * 8; idx += 128) {
      int r  = idx >> 3;
      int c4 = (idx & 7) << 2;
      long long ro = rowOff[r];
      int gk = k0 + c4;
      float v0 = 0.f, v1 = 0.f, v2 = 0.f, v3 = 0.f;
      if (ro >= 0 && gk < K) {
        if (gk + 4 <= K) {
          float t4[4];
          __builtin_memcpy(t4, A + ro + gk, 16);
          v0 = t4[0]; v1 = t4[1]; v2 = t4[2]; v3 = t4[3];
        } else {
          v0 = A[ro + gk];
          if (gk + 1 < K) v1 = A[ro + gk + 1];
          if (gk + 2 < K) v2 = A[ro + gk + 2];
        }
      }
      v4h h4 = { (_Float16)v0, (_Float16)v1, (_Float16)v2, (_Float16)v3 };
      *(v4h*)&As[r][c4] = h4;
    }
    // ---- stage B tile transposed: 32(k) x 32(n)
    if (sBn == 1) {                       // contiguous along n: b128 global loads
      for (int idx = tid; idx < BK * 8; idx += 128) {
        int k  = idx >> 3;
        int c4 = (idx & 7) << 2;
        int gk = k0 + k, gn = colBase + c4;
        float v0 = 0.f, v1 = 0.f, v2 = 0.f, v3 = 0.f;
        if (gk < K && gn < N) {
          if (gn + 4 <= N) {
            float t4[4];
            __builtin_memcpy(t4, B + (long long)gk * sBk + gn, 16);
            v0 = t4[0]; v1 = t4[1]; v2 = t4[2]; v3 = t4[3];
          } else {
            v0 = B[(long long)gk * sBk + gn];
            if (gn + 1 < N) v1 = B[(long long)gk * sBk + gn + 1];
            if (gn + 2 < N) v2 = B[(long long)gk * sBk + gn + 2];
          }
        }
        Bs[c4 + 0][k] = (_Float16)v0;
        Bs[c4 + 1][k] = (_Float16)v1;
        Bs[c4 + 2][k] = (_Float16)v2;
        Bs[c4 + 3][k] = (_Float16)v3;
      }
    } else {                              // strided B (conv OIH weight views)
      for (int idx = tid; idx < BK * 8; idx += 128) {
        int k  = idx >> 3;
        int c4 = (idx & 7) << 2;
        int gk = k0 + k;
#pragma unroll
        for (int j = 0; j < 4; ++j) {
          int gn = colBase + c4 + j;
          float v = 0.f;
          if (gk < K && gn < N) v = B[(long long)gk * sBk + (long long)gn * sBn];
          Bs[c4 + j][k] = (_Float16)v;
        }
      }
    }
    __syncthreads();

    // ---- fragments: two b128 LDS loads each (fragment-native layouts)
    const _Float16* ar = &As[wave * 16 + mlane][hi * 16];
    v8h a0 = *(const v8h*)ar;
    v8h a1 = *(const v8h*)(ar + 8);
    v16h af = __builtin_shufflevector(a0, a1, 0,1,2,3,4,5,6,7,8,9,10,11,12,13,14,15);

    const _Float16* br0 = &Bs[mlane][hi * 16];
    v8h b00 = *(const v8h*)br0;
    v8h b01 = *(const v8h*)(br0 + 8);
    v16h bf0 = __builtin_shufflevector(b00, b01, 0,1,2,3,4,5,6,7,8,9,10,11,12,13,14,15);

    const _Float16* br1 = &Bs[16 + mlane][hi * 16];
    v8h b10 = *(const v8h*)br1;
    v8h b11 = *(const v8h*)(br1 + 8);
    v16h bf1 = __builtin_shufflevector(b10, b11, 0,1,2,3,4,5,6,7,8,9,10,11,12,13,14,15);

    acc0 = __builtin_amdgcn_wmma_f32_16x16x32_f16(false, af, false, bf0,
                                                  (short)0, acc0, false, false);
    acc1 = __builtin_amdgcn_wmma_f32_16x16x32_f16(false, af, false, bf1,
                                                  (short)0, acc1, false, false);
    __syncthreads();
  }

  // ---- epilogue
#pragma unroll
  for (int nx = 0; nx < 2; ++nx) {
    int ncol = colBase + nx * 16 + mlane;
    if (ncol >= N) continue;
    float bv = (epi && bias) ? bias[ncol] : 0.f;
#pragma unroll
    for (int j = 0; j < 8; ++j) {
      int m = rowBase + wave * 16 + (hi << 3) + j;
      if (m < M) {
        long long ci = (long long)m * ldC + ncol;
        float v = nx ? acc1[j] : acc0[j];
        if (beta) v += C[ci];
        if (epi) {
          v += bv;
          if (act == 1)      v = v > 0.f ? v : 0.f;
          else if (act == 2) v = v > 0.f ? v : (__expf(v) - 1.f);
        }
        C[ci] = v;
      }
    }
  }
}

// ---------------- elementwise / fill ----------------
__global__ void fill_f32(float* p, float v, long long n) {
  long long i = blockIdx.x * 256LL + threadIdx.x;
  if (i < n) p[i] = v;
}
__global__ void fill_u32(unsigned* p, unsigned v, long long n) {
  long long i = blockIdx.x * 256LL + threadIdx.x;
  if (i < n) p[i] = v;
}
__global__ void bias_act_k(float* x, const float* __restrict__ bias,
                           long long total, int fdim, int act) {
  long long i = blockIdx.x * 256LL + threadIdx.x;
  if (i >= total) return;
  int f = (int)(i % fdim);
  float v = x[i] + bias[f];
  if (act == 1)      v = v > 0.f ? v : 0.f;
  else if (act == 2) v = v > 0.f ? v : (__expf(v) - 1.f);
  x[i] = v;
}

// ---------------- GAT attention kernels ----------------
__global__ void node_scores(const float* __restrict__ h, const float* __restrict__ att,
                            float* __restrict__ s, int n, int heads, int dout) {
  long long i = blockIdx.x * 256LL + threadIdx.x;
  if (i >= (long long)n * heads) return;
  int node = (int)(i / heads), hd = (int)(i % heads);
  const float* hp = h + (long long)node * heads * dout + (long long)hd * dout;
  const float* ap = att + (long long)hd * dout;
  float acc = 0.f;
  for (int d = 0; d < dout; ++d) acc += hp[d] * ap[d];
  s[i] = acc;
}

__global__ void edge_alpha(const int* __restrict__ srcA, const int* __restrict__ dstA,
                           int E, int n,
                           const float* __restrict__ ssrc, const float* __restrict__ sdst,
                           const float* __restrict__ se, int heads,
                           float* __restrict__ alpha, unsigned* __restrict__ amax) {
  long long i = blockIdx.x * 256LL + threadIdx.x;
  long long total = (long long)(E + n) * heads;
  if (i >= total) return;
  long long e = i / heads; int hd = (int)(i % heads);
  int s = (e < E) ? srcA[e] : (int)(e - E);
  int d = (e < E) ? dstA[e] : (int)(e - E);
  float a = ssrc[(long long)s * heads + hd] + sdst[(long long)d * heads + hd];
  if (se) a += se[i];
  a = a > 0.f ? a : 0.2f * a;
  alpha[i] = a;
  atomicMax(&amax[(long long)d * heads + hd], ord_map(a));
}

__global__ void edge_exp(const int* __restrict__ dstA, int E, int n,
                         const float* __restrict__ alpha, const unsigned* __restrict__ amax,
                         float* __restrict__ ex, float* __restrict__ denom, int heads) {
  long long i = blockIdx.x * 256LL + threadIdx.x;
  long long total = (long long)(E + n) * heads;
  if (i >= total) return;
  long long e = i / heads; int hd = (int)(i % heads);
  int d = (e < E) ? dstA[e] : (int)(e - E);
  float m = ord_unmap(amax[(long long)d * heads + hd]);
  float v = __expf(alpha[i] - m);
  ex[i] = v;
  atomicAdd(&denom[(long long)d * heads + hd], v);
}

__global__ void edge_aggregate(const int* __restrict__ srcA, const int* __restrict__ dstA,
                               int E, int n, const float* __restrict__ h,
                               const float* __restrict__ ex, const float* __restrict__ denom,
                               float* __restrict__ out, int heads, int dout) {
  long long i = blockIdx.x * 256LL + threadIdx.x;
  int F = heads * dout;
  long long total = (long long)(E + n) * F;
  if (i >= total) return;
  int f = (int)(i % F);
  long long e = i / F;
  int hd = f / dout;
  int s = (e < E) ? srcA[e] : (int)(e - E);
  int d = (e < E) ? dstA[e] : (int)(e - E);
  float w = ex[e * heads + hd] / (denom[(long long)d * heads + hd] + 1e-16f);
  atomicAdd(&out[(long long)d * F + f], h[(long long)s * F + f] * w);
}

// ---------------- edge-attr attention precompute ----------------
__global__ void edge_attr_sums(const float* __restrict__ ea, float* __restrict__ sum,
                               int E, int C) {
  long long i = blockIdx.x * 256LL + threadIdx.x;
  if (i >= (long long)E * C) return;
  atomicAdd(&sum[i % C], ea[i]);
}
__global__ void finalize_me(const float* __restrict__ We, const float* __restrict__ ae,
                            const float* __restrict__ sums, float* __restrict__ Me,
                            float* __restrict__ mean, int C, int heads, int dout, float invE) {
  int i = threadIdx.x;
  if (i < C * heads) {
    int c = i / heads, hd = i % heads;
    float acc = 0.f;
    for (int d = 0; d < dout; ++d)
      acc += We[(long long)c * heads * dout + (long long)hd * dout + d] * ae[(long long)hd * dout + d];
    Me[i] = acc;
  }
  if (i < C) mean[i] = sums[i] * invE;
}
__global__ void edge_se(const float* __restrict__ ea, const float* __restrict__ Me,
                        const float* __restrict__ mean, int E, int n, int C, int heads,
                        float* __restrict__ se) {
  long long i = blockIdx.x * 256LL + threadIdx.x;
  long long total = (long long)(E + n) * heads;
  if (i >= total) return;
  long long e = i / heads; int hd = (int)(i % heads);
  float acc = 0.f;
  for (int c = 0; c < C; ++c) {
    float v = (e < E) ? ea[e * C + c] : mean[c];
    acc += v * Me[c * heads + hd];
  }
  se[i] = acc;
}

// ---------------- pooling / gather ----------------
__global__ void seg_pool_max(const float* __restrict__ x, const int* __restrict__ batch,
                             unsigned* __restrict__ pool, int n, int fdim) {
  long long i = blockIdx.x * 256LL + threadIdx.x;
  if (i >= (long long)n * fdim) return;
  int node = (int)(i / fdim), f = (int)(i % fdim);
  atomicMax(&pool[(long long)batch[node] * fdim + f], ord_map(x[i]));
}
__global__ void unmap_k(const unsigned* __restrict__ in, float* __restrict__ outp, long long n) {
  long long i = blockIdx.x * 256LL + threadIdx.x;
  if (i < n) outp[i] = ord_unmap(in[i]);
}
__global__ void embed_gather(const int* __restrict__ tok, const float* __restrict__ emb,
                             float* __restrict__ outp, long long rows, int dim) {
  long long i = blockIdx.x * 256LL + threadIdx.x;
  if (i >= rows * dim) return;
  long long r = i / dim; int c = (int)(i % dim);
  outp[i] = emb[(long long)tok[r] * dim + c];
}
__global__ void time_max(const float* __restrict__ x, float* __restrict__ outp,
                         int Bb, int P, int Cc) {
  int i = blockIdx.x * 256 + threadIdx.x;
  if (i >= Bb * Cc) return;
  int b = i / Cc, c = i % Cc;
  float m = -3.0e38f;
  for (int p = 0; p < P; ++p) {
    float v = x[((long long)b * P + p) * Cc + c];
    m = v > m ? v : m;
  }
  outp[(long long)b * Cc + c] = m;
}

// ---------------- host helpers ----------------
static inline int cdiv(long long a, int b) { return (int)((a + b - 1) / b); }

static void launch_gemm(hipStream_t s, const float* A, int ldA, int Q, long long S,
                        const float* B, long long sBk, long long sBn,
                        float* C, int ldC, int M, int N, int K,
                        const float* bias, int act, int beta, int epi) {
  dim3 g((N + BN - 1) / BN, (M + BM - 1) / BM), b(128);
  gemm_wmma<<<g, b, 0, s>>>(A, ldA, Q, S, B, sBk, sBn, C, ldC, M, N, K, bias, act, beta, epi);
}

static void run_gat(hipStream_t st, const float* h, float* out,
                    const int* src, const int* dst, int E, int n, int heads, int dout,
                    const float* att_src, const float* att_dst, const float* bias, int act,
                    const float* se, float* ssrc, float* sdst, unsigned* amax, float* denom,
                    float* alpha, float* ex) {
  long long nh = (long long)n * heads;
  node_scores<<<cdiv(nh, 256), 256, 0, st>>>(h, att_src, ssrc, n, heads, dout);
  node_scores<<<cdiv(nh, 256), 256, 0, st>>>(h, att_dst, sdst, n, heads, dout);
  fill_u32<<<cdiv(nh, 256), 256, 0, st>>>(amax, 0u, nh);
  fill_f32<<<cdiv(nh, 256), 256, 0, st>>>(denom, 0.f, nh);
  long long eh = (long long)(E + n) * heads;
  edge_alpha<<<cdiv(eh, 256), 256, 0, st>>>(src, dst, E, n, ssrc, sdst, se, heads, alpha, amax);
  edge_exp<<<cdiv(eh, 256), 256, 0, st>>>(dst, E, n, alpha, amax, ex, denom, heads);
  long long of = (long long)n * heads * dout;
  fill_f32<<<cdiv(of, 256), 256, 0, st>>>(out, 0.f, of);
  long long tot = (long long)(E + n) * heads * dout;
  edge_aggregate<<<cdiv(tot, 256), 256, 0, st>>>(src, dst, E, n, h, ex, denom, out, heads, dout);
  bias_act_k<<<cdiv(of, 256), 256, 0, st>>>(out, bias, of, heads * dout, act);
}

static void run_pool(hipStream_t st, const float* x, const int* batch,
                     unsigned* poolu, float* poolf, int n, int fdim, int nb) {
  long long t1 = (long long)nb * fdim;
  fill_u32<<<cdiv(t1, 256), 256, 0, st>>>(poolu, 0u, t1);
  long long t2 = (long long)n * fdim;
  seg_pool_max<<<cdiv(t2, 256), 256, 0, st>>>(x, batch, poolu, n, fdim);
  unmap_k<<<cdiv(t1, 256), 256, 0, st>>>(poolu, poolf, t1);
}

// ---------------- entry point ----------------
extern "C" void kernel_launch(void* const* d_in, const int* in_sizes, int n_in,
                              void* d_out, int out_size, void* d_ws, size_t ws_size,
                              hipStream_t stream) {
  (void)in_sizes; (void)n_in; (void)out_size; (void)ws_size;
  const int N_D = 50000, E_D = 200000, N_A = 50000, E_A = 200000, Bb = 256, L = 1000;

  const float* drug_x  = (const float*)d_in[0];
  const int*   d_ei    = (const int*)d_in[1];
  const int*   d_batch = (const int*)d_in[2];
  const int*   tokens  = (const int*)d_in[3];
  const float* a_x     = (const float*)d_in[4];
  const int*   a_ei    = (const int*)d_in[5];
  const float* a_eatt  = (const float*)d_in[6];
  const int*   a_batch = (const int*)d_in[7];
  const float* gat1_w = (const float*)d_in[8];
  const float* gat1_as = (const float*)d_in[9];
  const float* gat1_ad = (const float*)d_in[10];
  const float* gat1_b = (const float*)d_in[11];
  const float* gat2_w = (const float*)d_in[12];
  const float* gat2_as = (const float*)d_in[13];
  const float* gat2_ad = (const float*)d_in[14];
  const float* gat2_b = (const float*)d_in[15];
  const float* fcg1_w = (const float*)d_in[16];
  const float* fcg1_b = (const float*)d_in[17];
  const float* emb_w  = (const float*)d_in[18];
  const float* conv1_w = (const float*)d_in[19];
  const float* conv1_b = (const float*)d_in[20];
  const float* conv2_w = (const float*)d_in[21];
  const float* conv2_b = (const float*)d_in[22];
  const float* fcxt_w = (const float*)d_in[23];
  const float* fcxt_b = (const float*)d_in[24];
  const float* ag1_w = (const float*)d_in[25];
  const float* ag1_as = (const float*)d_in[26];
  const float* ag1_ad = (const float*)d_in[27];
  const float* ag1_le = (const float*)d_in[28];
  const float* ag1_ae = (const float*)d_in[29];
  const float* ag1_b = (const float*)d_in[30];
  const float* ag2_w = (const float*)d_in[31];
  const float* ag2_as = (const float*)d_in[32];
  const float* ag2_ad = (const float*)d_in[33];
  const float* ag2_le = (const float*)d_in[34];
  const float* ag2_ae = (const float*)d_in[35];
  const float* ag2_b = (const float*)d_in[36];
  const float* afc1_w = (const float*)d_in[37];
  const float* afc1_b = (const float*)d_in[38];
  const float* fc1_w = (const float*)d_in[39];
  const float* fc1_b = (const float*)d_in[40];
  const float* fc2_w = (const float*)d_in[41];
  const float* fc2_b = (const float*)d_in[42];
  const float* out_w = (const float*)d_in[43];
  const float* out_b = (const float*)d_in[44];
  float* out = (float*)d_out;

  float* ws = (float*)d_ws;
  // persistent region
  float* XC = ws;                 // [256,384] fused concat
  float* F1 = ws + 98304;         // [256,1024]
  float* F2 = ws + 360448;        // [256,256]
  float* S  = ws + 425984;        // reusable scratch arena

  // ================= drug GAT branch =================
  {
    float* h1 = S;                      // [50000,390]
    float* x1 = S + 19500000;           // [50000,390]
    float* sm = S + 39000000;
    float* ssrc = sm;                   float* sdst = sm + 250000;
    unsigned* amax = (unsigned*)(sm + 500000);
    float* denom = sm + 750000;
    float* alpha = sm + 1000000;        float* ex = sm + 2250000;
    unsigned* poolu = (unsigned*)(S + 42500000);
    float* poolf = S + 42532768;
    const int* src = d_ei; const int* dst = d_ei + E_D;

    launch_gemm(stream, drug_x, 78, 0, 0, gat1_w, 390, 1, h1, 390, N_D, 390, 78,
                nullptr, 0, 0, 0);
    run_gat(stream, h1, x1, src, dst, E_D, N_D, 5, 78, gat1_as, gat1_ad, gat1_b, 2,
            nullptr, ssrc, sdst, amax, denom, alpha, ex);

    float* h2 = S;                      // [50000,128]
    float* x2 = S + 6400000;            // [50000,128]
    launch_gemm(stream, x1, 390, 0, 0, gat2_w, 128, 1, h2, 128, N_D, 128, 390,
                nullptr, 0, 0, 0);
    run_gat(stream, h2, x2, src, dst, E_D, N_D, 1, 128, gat2_as, gat2_ad, gat2_b, 1,
            nullptr, ssrc, sdst, amax, denom, alpha, ex);

    run_pool(stream, x2, d_batch, poolu, poolf, N_D, 128, Bb);
    launch_gemm(stream, poolf, 128, 0, 0, fcg1_w, 128, 1, XC, 384, Bb, 128, 128,
                fcg1_b, 1, 0, 1);
  }

  // ================= protein CNN branch =================
  {
    float* emb = S;                     // [256*1000,128]
    float* c1  = S + 32768000;          // [256*996,64]
    float* c2  = S + 49086464;          // [256*994,32]
    float* xtp = S + 57229312;          // [256,32]
    long long rows = (long long)Bb * L;
    embed_gather<<<cdiv(rows * 128, 256), 256, 0, stream>>>(tokens, emb_w, emb, rows, 128);

    for (int t = 0; t < 5; ++t)
      launch_gemm(stream, emb + (long long)t * 128, 128, 996, (long long)L * 128,
                  conv1_w + t, 5, 640, c1, 64, Bb * 996, 64, 128,
                  conv1_b, 1, (t > 0), (t == 4));
    for (int t = 0; t < 3; ++t)
      launch_gemm(stream, c1 + (long long)t * 64, 64, 994, (long long)996 * 64,
                  conv2_w + t, 3, 192, c2, 32, Bb * 994, 32, 64,
                  conv2_b, 1, (t > 0), (t == 2));

    time_max<<<cdiv(Bb * 32, 256), 256, 0, stream>>>(c2, xtp, Bb, 994, 32);
    launch_gemm(stream, xtp, 32, 0, 0, fcxt_w, 128, 1, XC + 128, 384, Bb, 128, 32,
                fcxt_b, 0, 0, 1);
  }

  // ================= a2h GAT branch (edge features) =================
  {
    float* ha = S;                      // [50000,320]
    float* xa = S + 16000000;           // [50000,320]
    float* sm = S + 32000000;
    float* ssrc = sm;                   float* sdst = sm + 250000;
    unsigned* amax = (unsigned*)(sm + 500000);
    float* denom = sm + 750000;
    float* alpha = sm + 1000000;        float* ex = sm + 2250000;
    float* se = sm + 3500000;           // [(E+N)*heads]
    float* Me = sm + 4750000;           float* meanv = sm + 4750064;
    float* sums = sm + 4750128;
    unsigned* poolu = (unsigned*)(sm + 4750192);
    float* poolf = sm + 4750192 + 32768;
    const int* src = a_ei; const int* dst = a_ei + E_A;

    launch_gemm(stream, a_x, 24, 0, 0, ag1_w, 320, 1, ha, 320, N_A, 320, 24,
                nullptr, 0, 0, 0);
    fill_f32<<<1, 64, 0, stream>>>(sums, 0.f, 8);
    edge_attr_sums<<<cdiv((long long)E_A * 5, 256), 256, 0, stream>>>(a_eatt, sums, E_A, 5);
    finalize_me<<<1, 64, 0, stream>>>(ag1_le, ag1_ae, sums, Me, meanv, 5, 5, 64, 1.f / E_A);
    edge_se<<<cdiv((long long)(E_A + N_A) * 5, 256), 256, 0, stream>>>(
        a_eatt, Me, meanv, E_A, N_A, 5, 5, se);
    run_gat(stream, ha, xa, src, dst, E_A, N_A, 5, 64, ag1_as, ag1_ad, ag1_b, 2,
            se, ssrc, sdst, amax, denom, alpha, ex);

    float* h2 = S;                      // [50000,128]
    float* x2 = S + 6400000;            // [50000,128]
    launch_gemm(stream, xa, 320, 0, 0, ag2_w, 128, 1, h2, 128, N_A, 128, 320,
                nullptr, 0, 0, 0);
    finalize_me<<<1, 64, 0, stream>>>(ag2_le, ag2_ae, sums, Me, meanv, 5, 1, 128, 1.f / E_A);
    edge_se<<<cdiv((long long)(E_A + N_A), 256), 256, 0, stream>>>(
        a_eatt, Me, meanv, E_A, N_A, 5, 1, se);
    run_gat(stream, h2, x2, src, dst, E_A, N_A, 1, 128, ag2_as, ag2_ad, ag2_b, 1,
            se, ssrc, sdst, amax, denom, alpha, ex);

    run_pool(stream, x2, a_batch, poolu, poolf, N_A, 128, Bb);
    launch_gemm(stream, poolf, 128, 0, 0, afc1_w, 128, 1, XC + 256, 384, Bb, 128, 128,
                afc1_b, 1, 0, 1);
  }

  // ================= fusion head =================
  launch_gemm(stream, XC, 384, 0, 0, fc1_w, 1024, 1, F1, 1024, Bb, 1024, 384,
              fc1_b, 1, 0, 1);
  launch_gemm(stream, F1, 1024, 0, 0, fc2_w, 256, 1, F2, 256, Bb, 256, 1024,
              fc2_b, 1, 0, 1);
  launch_gemm(stream, F2, 256, 0, 0, out_w, 1, 1, out, 1, Bb, 1, 256,
              out_b, 0, 0, 1);
}
